// StackedMinGRU_67259187855490
// MI455X (gfx1250) — matrix operand order repaired and checked by reading, not compiled, not run on hardware
//
#include <hip/hip_runtime.h>
#include <hip/hip_bf16.h>

#define NLAYERS 4
#define HID 512
#define BATCH 32
#define SEQ 2048
#define MROWS (BATCH*SEQ)      // 65536
#define NCH 16                 // scan chunks
#define CT (SEQ/NCH)           // 128
#define BH (BATCH*HID)         // 16384
#define LDT 40                 // LDS row stride in bf16 halves (32 + 8 pad)

typedef __attribute__((ext_vector_type(16))) __bf16 v16bf;
typedef __attribute__((ext_vector_type(8)))  float  v8f;
typedef __attribute__((ext_vector_type(4)))  float  f4;

__device__ __forceinline__ float sigmoidf_(float x){ return 1.0f/(1.0f+__expf(-x)); }
__device__ __forceinline__ float softplusf_(float x){ return (x>20.f)? x : log1pf(__expf(x)); }
__device__ __forceinline__ float gfun_(float x){ return (x>=0.f)? (x+0.5f) : sigmoidf_(x); }

// async global->LDS copy of 16B per lane (CDNA5 GLOBAL_LOAD_ASYNC_TO_LDS_B128, GVS mode)
#define ASYNC_LDS_B128(ldsaddr, voff, sbase)                                  \
  asm volatile("global_load_async_to_lds_b128 %0, %1, %2"                     \
               :: "v"(ldsaddr), "v"(voff), "s"(sbase) : "memory")

// ---- weight prep: fp32 (k-major) -> bf16 transposed (n-major) ----
__global__ __launch_bounds__(256)
void wprep_kernel(const float* __restrict__ Wz, const float* __restrict__ Wh,
                  __bf16* __restrict__ WzT, __bf16* __restrict__ WhT){
  int idx = blockIdx.x*256 + threadIdx.x;          // l*H*H + n*H + k
  if (idx >= NLAYERS*HID*HID) return;
  int l   = idx >> 18;
  int rem = idx & (HID*HID - 1);
  int n   = rem >> 9;
  int k   = rem & (HID - 1);
  int src = l*HID*HID + k*HID + n;
  WzT[idx] = (__bf16)Wz[src];
  WhT[idx] = (__bf16)Wh[src];
}

// ---- activation convert: fp32 -> bf16 (for layer-0 GEMM A operand) ----
__global__ __launch_bounds__(256)
void xcvt_kernel(const float* __restrict__ in, __bf16* __restrict__ out){
  size_t i = ((size_t)blockIdx.x*256 + threadIdx.x) * 4;
  f4 v = *(const f4*)(in + i);
  union { __bf16 h[4]; unsigned long long q; } u;
  #pragma unroll
  for (int e=0;e<4;e++) u.h[e] = (__bf16)v[e];
  *(unsigned long long*)(out + i) = u.q;
}

// ---- dual GEMM: Ck = A@Wz + bz, Cp = A@Wh + bh ----
// A bf16 row-major, weights bf16 n-major. Double-buffered LDS fed by
// GLOBAL_LOAD_ASYNC_TO_LDS_B128, counted s_wait_asynccnt for pipelining.
__global__ __launch_bounds__(256)
void gemm2_kernel(const __bf16* __restrict__ A,
                  const __bf16* __restrict__ BzT, const __bf16* __restrict__ BhT,
                  const float* __restrict__ bz, const float* __restrict__ bh,
                  float* __restrict__ Ck, float* __restrict__ Cp){
  __shared__ __align__(16) __bf16 sA [2][128*LDT];
  __shared__ __align__(16) __bf16 sBz[2][128*LDT];
  __shared__ __align__(16) __bf16 sBh[2][128*LDT];

  const int tid  = threadIdx.x;
  const int lane = tid & 31;
  const int wave = tid >> 5;
  const int wm   = wave & 3;     // M quadrant (32 rows each)
  const int wn   = wave >> 2;    // 0..1 -> N half (64 cols)
  const int lrow = lane & 15;
  const int hi   = lane >> 4;

  const int m0 = blockIdx.x * 128;
  const int n0 = blockIdx.y * 128;

  // SGPR bases (uniform), per-lane byte offsets for async staging
  const __bf16* aBase  = A   + (size_t)m0*HID;
  const __bf16* bzBase = BzT + (size_t)n0*HID;
  const __bf16* bhBase = BhT + (size_t)n0*HID;
  const int r  = tid >> 1;
  const int cp = tid & 1;
  const unsigned rowoff = (unsigned)(r*HID*2 + cp*32);       // bytes into tile row set
  const unsigned ldsoff = (unsigned)(r*LDT*2 + cp*32);       // bytes into LDS tile

  v8f acck[2][4];
  v8f accp[2][4];
  #pragma unroll
  for (int i=0;i<2;i++)
    #pragma unroll
    for (int j=0;j<4;j++){
      acck[i][j] = (v8f){0.f,0.f,0.f,0.f,0.f,0.f,0.f,0.f};
      accp[i][j] = (v8f){0.f,0.f,0.f,0.f,0.f,0.f,0.f,0.f};
    }

  // issue 6 async b128 copies (A/Bz/Bh tiles) for K-step kc into buffer bsel
  auto issue = [&](int kc, int bsel){
    const unsigned go = rowoff + (unsigned)kc*64;            // global byte offset
    unsigned la = (unsigned)(uintptr_t)&sA [bsel][0] + ldsoff;
    unsigned lz = (unsigned)(uintptr_t)&sBz[bsel][0] + ldsoff;
    unsigned lh = (unsigned)(uintptr_t)&sBh[bsel][0] + ldsoff;
    ASYNC_LDS_B128(la,      go,      aBase);
    ASYNC_LDS_B128(la + 16, go + 16, aBase);
    ASYNC_LDS_B128(lz,      go,      bzBase);
    ASYNC_LDS_B128(lz + 16, go + 16, bzBase);
    ASYNC_LDS_B128(lh,      go,      bhBase);
    ASYNC_LDS_B128(lh + 16, go + 16, bhBase);
  };

  issue(0, 0);
  for (int kc = 0; kc < 16; ++kc){
    const int cb = kc & 1;
    if (kc < 15){
      issue(kc + 1, cb ^ 1);
      asm volatile("s_wait_asynccnt 0x6" ::: "memory");  // current buffer retired
    } else {
      asm volatile("s_wait_asynccnt 0x0" ::: "memory");
    }
    __syncthreads();                                     // all waves staged

    // A fragments: lane = M row; ISA 16-bit A 16x32 VGPR layout
    v16bf afr[2];
    #pragma unroll
    for (int mi=0;mi<2;mi++){
      const __bf16* p = &sA[cb][(wm*32 + mi*16 + lrow)*LDT];
      f4* d = (f4*)&afr[mi];
      d[0] = *(const f4*)(p + hi*8);        // K = hi*8 .. +7
      d[1] = *(const f4*)(p + 16 + hi*8);   // K = 16+hi*8 .. +7
    }
    #pragma unroll
    for (int ni=0;ni<4;ni++){
      const int nn = wn*64 + ni*16 + lrow;
      v16bf bzf, bhf;
      {
        const __bf16* pz = &sBz[cb][nn*LDT + hi*8];
        f4* d = (f4*)&bzf; d[0]=*(const f4*)pz; d[1]=*(const f4*)(pz + 16);
        const __bf16* ph = &sBh[cb][nn*LDT + hi*8];
        f4* e = (f4*)&bhf; e[0]=*(const f4*)ph; e[1]=*(const f4*)(ph + 16);
      }
      #pragma unroll
      for (int mi=0;mi<2;mi++){
        acck[mi][ni] = __builtin_amdgcn_wmma_f32_16x16x32_bf16(
            false, afr[mi], false, bzf, (short)0, acck[mi][ni], false, false);
        accp[mi][ni] = __builtin_amdgcn_wmma_f32_16x16x32_bf16(
            false, afr[mi], false, bhf, (short)0, accp[mi][ni], false, false);
      }
    }
    __syncthreads();   // safe to overwrite this buffer two iterations later
  }

  // epilogue: C/D layout -> lane holds column n, VGPR r8 is row (hi*8 + r8)
  #pragma unroll
  for (int mi=0;mi<2;mi++){
    #pragma unroll
    for (int ni=0;ni<4;ni++){
      const int nn    = n0 + wn*64 + ni*16 + lrow;
      const int mbase = m0 + wm*32 + mi*16 + hi*8;
      const float bzv = bz[nn];
      const float bhv = bh[nn];
      #pragma unroll
      for (int r8=0;r8<8;r8++){
        size_t off = (size_t)(mbase + r8)*HID + nn;
        Ck[off] = acck[mi][ni][r8] + bzv;
        Cp[off] = accp[mi][ni][r8] + bhv;
      }
    }
  }
}

// ---- scan pass 1: per-chunk log-decay and zero-init response ----
__global__ __launch_bounds__(256)
void scan1_kernel(const float* __restrict__ K, const float* __restrict__ P,
                  float* __restrict__ logp, float* __restrict__ s){
  const int ch = blockIdx.x*256 + threadIdx.x;   // b*HID + h
  const int c  = blockIdx.y;
  const int b  = ch >> 9;
  const int hx = ch & (HID-1);
  size_t idx = ((size_t)(b*SEQ + c*CT))*HID + hx;
  float lp = 0.f, sacc = 0.f;
  for (int t=0;t<CT;t++){
    const float kv = K[idx];
    const float pv = P[idx];
    const float a  = sigmoidf_(-kv);             // 1 - z
    const float bb = sigmoidf_(kv) * gfun_(pv);  // z * g(pre)
    sacc = fmaf(a, sacc, bb);
    lp  -= softplusf_(kv);                       // += log(1-z), stable
    idx += HID;
  }
  logp[c*BH + ch] = lp;
  s   [c*BH + ch] = sacc;
}

// ---- scan pass 2: serial combine across 16 chunks; emits h[T-1] (hiddens out) ----
__global__ __launch_bounds__(256)
void scan2_kernel(const float* __restrict__ h0l,
                  const float* __restrict__ logp, const float* __restrict__ s,
                  float* __restrict__ hin, float* __restrict__ hlast){
  const int ch = blockIdx.x*256 + threadIdx.x;
  float h = gfun_(h0l[ch]);
  #pragma unroll
  for (int c=0;c<NCH;c++){
    hin[c*BH + ch] = h;
    h = fmaf(__expf(logp[c*BH + ch]), h, s[c*BH + ch]);
  }
  hlast[ch] = h;
}

// ---- scan pass 3: replay each chunk with correct initial state ----
__global__ __launch_bounds__(256)
void scan3_kernel(const float* __restrict__ K, const float* __restrict__ P,
                  const float* __restrict__ hin, float* __restrict__ H){
  const int ch = blockIdx.x*256 + threadIdx.x;
  const int c  = blockIdx.y;
  const int b  = ch >> 9;
  const int hx = ch & (HID-1);
  size_t idx = ((size_t)(b*SEQ + c*CT))*HID + hx;
  float h = hin[c*BH + ch];
  for (int t=0;t<CT;t++){
    const float kv = K[idx];
    const float pv = P[idx];
    const float a  = sigmoidf_(-kv);
    const float bb = sigmoidf_(kv) * gfun_(pv);
    h = fmaf(a, h, bb);
    H[idx] = h;
    idx += HID;
  }
}

// ---- layernorm + residual: one wave32 per 512-wide row ----
// writes fp32 residual stream (out) and bf16 activation stream (outbf) for next GEMM
__global__ __launch_bounds__(256)
void ln_kernel(const float* __restrict__ H, const float* __restrict__ cur,
               const float* __restrict__ gamma, const float* __restrict__ beta,
               float* __restrict__ out, __bf16* __restrict__ outbf){
  const int wave = threadIdx.x >> 5;
  const int lane = threadIdx.x & 31;
  const size_t row  = (size_t)blockIdx.x*8 + wave;
  const size_t base = row*HID + lane*16;
  f4 v[4];
  #pragma unroll
  for (int j=0;j<4;j++) v[j] = *(const f4*)(H + base + j*4);
  float sum = 0.f;
  #pragma unroll
  for (int j=0;j<4;j++) sum += v[j][0]+v[j][1]+v[j][2]+v[j][3];
  #pragma unroll
  for (int m=16;m>0;m>>=1) sum += __shfl_xor(sum, m, 32);
  const float mean = sum * (1.0f/HID);
  float sq = 0.f;
  #pragma unroll
  for (int j=0;j<4;j++){
    #pragma unroll
    for (int e=0;e<4;e++){ const float d = v[j][e]-mean; sq = fmaf(d,d,sq); }
  }
  #pragma unroll
  for (int m=16;m>0;m>>=1) sq += __shfl_xor(sq, m, 32);
  const float rstd = rsqrtf(sq*(1.0f/HID) + 1e-5f);
  const int nb = lane*16;
  union { __bf16 h[16]; f4 v[2]; } ub;
  #pragma unroll
  for (int j=0;j<4;j++){
    f4 g  = *(const f4*)(gamma + nb + j*4);
    f4 bb = *(const f4*)(beta  + nb + j*4);
    f4 cc = *(const f4*)(cur + base + j*4);
    f4 o;
    #pragma unroll
    for (int e=0;e<4;e++){
      o[e] = (v[j][e]-mean)*rstd*g[e] + bb[e] + cc[e];
      ub.h[j*4+e] = (__bf16)o[e];
    }
    *(f4*)(out + base + j*4) = o;
  }
  ((f4*)(outbf + base))[0] = ub.v[0];
  ((f4*)(outbf + base))[1] = ub.v[1];
}

extern "C" void kernel_launch(void* const* d_in, const int* in_sizes, int n_in,
                              void* d_out, int out_size, void* d_ws, size_t ws_size,
                              hipStream_t stream) {
  const float* x     = (const float*)d_in[0];
  const float* h0    = (const float*)d_in[1];
  const float* Wz    = (const float*)d_in[2];
  const float* bz    = (const float*)d_in[3];
  const float* Wh    = (const float*)d_in[4];
  const float* bh    = (const float*)d_in[5];
  const float* gamma = (const float*)d_in[6];
  const float* beta  = (const float*)d_in[7];
  float* out = (float*)d_out;

  const size_t big = (size_t)MROWS * HID;          // 33.5M floats = 128 MB
  float* buf0 = (float*)d_ws;                      // k / cur (ping)
  float* buf1 = buf0 + big;                        // k / cur (pong)
  float* bufP = buf0 + 2*big;                      // pre
  float* bufH = buf0 + 3*big;                      // h
  char*  extra = (char*)(buf0 + 4*big);
  __bf16* curbf = (__bf16*)extra;                          // bf16 activations (64MB)
  __bf16* WzT = curbf + big;                               // L*H*H bf16
  __bf16* WhT = WzT + (size_t)NLAYERS*HID*HID;
  float*  logp = (float*)(WhT + (size_t)NLAYERS*HID*HID);  // NCH*BH
  float*  sbuf = logp + (size_t)NCH*BH;
  float*  hin  = sbuf + (size_t)NCH*BH;

  {
    const int tot = NLAYERS*HID*HID;
    wprep_kernel<<<(tot+255)/256, 256, 0, stream>>>(Wz, Wh, WzT, WhT);
    xcvt_kernel<<<(unsigned)(big/(256*4)), 256, 0, stream>>>(x, curbf);
  }

  const float* cur = x;
  for (int i=0;i<NLAYERS;i++){
    float* kb = (i & 1) ? buf1 : buf0;
    gemm2_kernel<<<dim3(MROWS/128, HID/128), 256, 0, stream>>>(
        curbf, WzT + (size_t)i*HID*HID, WhT + (size_t)i*HID*HID,
        bz + i*HID, bh + i*HID, kb, bufP);
    scan1_kernel<<<dim3(BH/256, NCH), 256, 0, stream>>>(kb, bufP, logp, sbuf);
    scan2_kernel<<<BH/256, 256, 0, stream>>>(h0 + (size_t)i*BH, logp, sbuf, hin,
                                             out + big + (size_t)i*BH);
    scan3_kernel<<<dim3(BH/256, NCH), 256, 0, stream>>>(kb, bufP, hin, bufH);
    float* op = (i == NLAYERS-1) ? out : kb;       // cur_out aliases dead k buffer
    ln_kernel<<<MROWS/8, 256, 0, stream>>>(bufH, cur, gamma + i*HID, beta + i*HID,
                                           op, curbf);
    cur = op;
  }
}
